// LogicExpert_51127290691883
// MI455X (gfx1250) — matrix operand order repaired and checked by reading, not compiled, not run on hardware
//
#include <hip/hip_runtime.h>
#include <hip/hip_bf16.h>

#define DEV __device__ __forceinline__

typedef __attribute__((ext_vector_type(16))) __bf16 v16bf;
typedef __attribute__((ext_vector_type(8)))  __bf16 v8bf;
typedef __attribute__((ext_vector_type(8)))  float  v8f;
typedef int v4i __attribute__((vector_size(4 * sizeof(int))));

static constexpr int BB   = 2;
static constexpr int SS   = 1024;
static constexpr int DD   = 768;
static constexpr int HH   = 12;
static constexpr int DHD  = 64;
static constexpr int DFFC = 3072;
static constexpr int CHC  = 64;
static constexpr int NTOK = BB * SS;   // 2048

#if __has_builtin(__builtin_amdgcn_global_load_async_to_lds_b128)
#define ASYNC_LDS 1
#else
#define ASYNC_LDS 0
#endif

// ---------------------------------------------------------------------------
// WMMA fragment helpers (bf16 16x16x32, contiguous K order k = 16*half + i, same for A and B)
// ---------------------------------------------------------------------------

// A 16x32 bf16: lane m (0..15)/m+16 holds row m; elems 0..15 = K half*16+i.
// rowbase = &A[row][kb], contiguous K.
DEV v16bf ld_a_frag(const __bf16* rowbase, int half) {
  v8bf x = *(const v8bf*)(rowbase + half * 16);
  v8bf y = *(const v8bf*)(rowbase + half * 16 + 8);
  v16bf a;
#pragma unroll
  for (int i = 0; i < 8; ++i) { a[i] = x[i]; a[8 + i] = y[i]; }
  return a;
}

// B 32x16 bf16: lanes 0..15 hold K=0..15 of column n, lanes 16..31 hold
// K=16..31.  colbase = &Bcol[n][kb + half*16], 16 contiguous K elements.
DEV v16bf ld_b_frag(const __bf16* colbase) {
  v8bf x = *(const v8bf*)(colbase);
  v8bf y = *(const v8bf*)(colbase + 8);
  v16bf b;
#pragma unroll
  for (int i = 0; i < 8; ++i) { b[i] = x[i]; b[8 + i] = y[i]; }
  return b;
}

DEV v8f wmma_bf16(v16bf a, v16bf b, v8f c) {
  return __builtin_amdgcn_wmma_f32_16x16x32_bf16(false, a, false, b,
                                                 (short)0, c, false, false);
}

DEV float gelu_f(float x) {
  return 0.5f * x * (1.0f + erff(x * 0.70710678118654752f));
}

// Stage one 16-byte chunk global -> LDS (async DMA when available)
DEV void stage_b128(const __bf16* gsrc, __bf16* ldst) {
#if ASYNC_LDS
  __builtin_amdgcn_global_load_async_to_lds_b128(
      (__attribute__((address_space(1))) v4i*)(unsigned long long)(uintptr_t)gsrc,
      (__attribute__((address_space(3))) v4i*)(unsigned int)(uintptr_t)ldst,
      0, 0);
#else
  *(uint4*)ldst = *(const uint4*)gsrc;
#endif
}

DEV void wait_async(int pending) {
#if ASYNC_LDS
  if (pending) asm volatile("s_wait_asynccnt 1" ::: "memory");
  else         asm volatile("s_wait_asynccnt 0" ::: "memory");
#else
  (void)pending;
#endif
}

// ---------------------------------------------------------------------------
// Elementwise cast / transpose-cast (weights to bf16, col-major "Wt")
// ---------------------------------------------------------------------------

__global__ __launch_bounds__(256) void transpose_cast_kernel(
    const float* __restrict__ src, __bf16* __restrict__ dst, int K, int N) {
  size_t idx = (size_t)blockIdx.x * 256 + threadIdx.x;  // over K*N
  int k = (int)(idx / N), n = (int)(idx % N);
  dst[(size_t)n * K + k] = (__bf16)src[idx];
}

__global__ __launch_bounds__(256) void cast_bf16_kernel(
    const float* __restrict__ src, __bf16* __restrict__ dst) {
  size_t idx = (size_t)blockIdx.x * 256 + threadIdx.x;
  dst[idx] = (__bf16)src[idx];
}

// ---------------------------------------------------------------------------
// Row LayerNorm (768 wide) -> bf16
// ---------------------------------------------------------------------------

__global__ __launch_bounds__(256) void ln_rows_bf16(
    const float* __restrict__ x, const float* __restrict__ g,
    const float* __restrict__ bta, __bf16* __restrict__ out) {
  __shared__ float sb[16];
  const int row = blockIdx.x, tid = threadIdx.x;
  const float* xr = x + (size_t)row * DD;
  float v[3];
#pragma unroll
  for (int t = 0; t < 3; ++t) v[t] = xr[tid + t * 256];
  float s = v[0] + v[1] + v[2];
#pragma unroll
  for (int m = 16; m >= 1; m >>= 1) s += __shfl_xor(s, m, 32);
  if ((tid & 31) == 0) sb[tid >> 5] = s;
  __syncthreads();
  float mean = 0.f;
#pragma unroll
  for (int w = 0; w < 8; ++w) mean += sb[w];
  mean *= (1.0f / DD);
  float q = 0.f;
#pragma unroll
  for (int t = 0; t < 3; ++t) { float d = v[t] - mean; q += d * d; }
#pragma unroll
  for (int m = 16; m >= 1; m >>= 1) q += __shfl_xor(q, m, 32);
  if ((tid & 31) == 0) sb[8 + (tid >> 5)] = q;
  __syncthreads();
  float var = 0.f;
#pragma unroll
  for (int w = 0; w < 8; ++w) var += sb[8 + w];
  var *= (1.0f / DD);
  const float inv = rsqrtf(var + 1e-5f);
  __bf16* orow = out + (size_t)row * DD;
#pragma unroll
  for (int t = 0; t < 3; ++t) {
    int c = tid + t * 256;
    orow[c] = (__bf16)((v[t] - mean) * inv * g[c] + bta[c]);
  }
}

// ---------------------------------------------------------------------------
// Generic bf16 WMMA GEMM: out = A[MxK] @ Wt[NxK]^T (+bias) with epilogue.
// block = 256 thr (8 waves), block tile 64x64, wave tile 16x32.
// B panel (64 rows x 32 K = 4KB) staged global->LDS (async DMA, double buf).
// ---------------------------------------------------------------------------

enum : int {
  EPI_BF16_BIAS = 0,   // bf16 out = acc + bias
  EPI_BF16_QSCALE,     // bf16 out = (acc + bias) * 0.125   (pre-scaled Q)
  EPI_BF16_VT,         // bf16 out -> V transposed to [B,H,DH,S]
  EPI_F32_BIAS_RES,    // f32 out = acc + bias + res
  EPI_F32_BIAS,        // f32 out = acc + bias
  EPI_F32_NONE,        // f32 out = acc
  EPI_BF16_GELU        // bf16 out = gelu(acc + bias)
};

template <int EPI>
__global__ __launch_bounds__(256) void gemm_bf16_kernel(
    const __bf16* __restrict__ A, const __bf16* __restrict__ Wt,
    const float* __restrict__ bias, const float* __restrict__ res,
    void* __restrict__ outp, int M, int N, int K) {
  __shared__ __bf16 bsm[2][64 * 32];   // double-buffered B panel (2 x 4KB)
  const int tid = threadIdx.x;
  const int lane = tid & 31, wave = tid >> 5;
  const int half = lane >> 4, ln = lane & 15;
  const int mtile = blockIdx.x * 64 + (wave & 3) * 16;
  const int ntile0 = blockIdx.y * 64;            // block's N base
  const int wn = (wave >> 2) * 32;               // wave's N offset in panel

  // staging map: thread -> one 16B chunk of the 64x32 panel
  const int sn = tid >> 2;                        // 0..63 panel row (N)
  const int sk = (tid & 3) * 8;                   // 0,8,16,24 (K elems)
  const __bf16* gsrc0 = Wt + (size_t)(ntile0 + sn) * K + sk;
  __bf16* lds_dst = &bsm[0][sn * 32 + sk];

  const __bf16* arow = A + (size_t)(mtile + ln) * K;
  // wave's B columns inside the LDS panel
  const int bcol0 = (wn + ln) * 32 + half * 16;
  const int bcol1 = (wn + 16 + ln) * 32 + half * 16;

  v8f acc0 = {}; v8f acc1 = {};

  stage_b128(gsrc0, lds_dst);                     // panel 0 -> buf 0
  int pb = 0;
  for (int kb = 0; kb < K; kb += 32, pb ^= 1) {
    const int more = (kb + 32 < K);
    if (more) stage_b128(gsrc0 + kb + 32, &bsm[pb ^ 1][sn * 32 + sk]);
    wait_async(more);
    __syncthreads();                              // buf[pb] ready everywhere

    v16bf a  = ld_a_frag(arow + kb, half);
    v16bf b0 = ld_b_frag(&bsm[pb][bcol0]);
    v16bf b1 = ld_b_frag(&bsm[pb][bcol1]);
    acc0 = wmma_bf16(a, b0, acc0);
    acc1 = wmma_bf16(a, b1, acc1);
    __syncthreads();                              // done reading before reuse
  }

#pragma unroll
  for (int r = 0; r < 8; ++r) {
    const int gm = mtile + half * 8 + r;
#pragma unroll
    for (int j = 0; j < 2; ++j) {
      const int gn = ntile0 + wn + j * 16 + ln;
      float v = j ? acc1[r] : acc0[r];
      if (EPI != EPI_F32_NONE) v += bias[gn];
      if (EPI == EPI_BF16_BIAS) {
        ((__bf16*)outp)[(size_t)gm * N + gn] = (__bf16)v;
      } else if (EPI == EPI_BF16_QSCALE) {
        ((__bf16*)outp)[(size_t)gm * N + gn] = (__bf16)(v * 0.125f);
      } else if (EPI == EPI_BF16_VT) {
        const int b = gm >> 10, s = gm & (SS - 1);
        const int h = gn >> 6,  d = gn & (DHD - 1);
        ((__bf16*)outp)[((size_t)((b * HH + h) * DHD + d) << 10) + s] = (__bf16)v;
      } else if (EPI == EPI_F32_BIAS_RES) {
        ((float*)outp)[(size_t)gm * N + gn] = v + res[(size_t)gm * N + gn];
      } else if (EPI == EPI_F32_BIAS || EPI == EPI_F32_NONE) {
        ((float*)outp)[(size_t)gm * N + gn] = v;
      } else if (EPI == EPI_BF16_GELU) {
        ((__bf16*)outp)[(size_t)gm * N + gn] = (__bf16)gelu_f(v);
      }
    }
  }
}

// ---------------------------------------------------------------------------
// Flash attention: one wave per (b, h, 16-query tile); Q pre-scaled by 1/8.
// Q,K: [B*S, D] bf16 (head-sliced rows);  Vt: [B,H,DH,S] bf16.
// ---------------------------------------------------------------------------

__global__ __launch_bounds__(256) void attn_kernel(
    const __bf16* __restrict__ Q, const __bf16* __restrict__ Km,
    const __bf16* __restrict__ Vt, __bf16* __restrict__ ctx) {
  __shared__ __bf16 pl[8 * 16 * 32];   // per-wave P tile (16x32)
  const int lane = threadIdx.x & 31, wave = threadIdx.x >> 5;
  const int half = lane >> 4, ln = lane & 15;
  const int task = blockIdx.x * 8 + wave;            // B*H*(S/16) = 1536
  const int b  = task / (HH * (SS / 16));
  const int rm = task % (HH * (SS / 16));
  const int h  = rm / (SS / 16);
  const int qt = rm % (SS / 16);

  const __bf16* qrow = Q + (size_t)(b * SS + qt * 16 + ln) * DD + h * DHD;
  const v16bf a0 = ld_a_frag(qrow, half);        // dh 0..31
  const v16bf a1 = ld_a_frag(qrow + 32, half);   // dh 32..63

  const __bf16* kbase = Km + (size_t)(b * SS) * DD + h * DHD + half * 16;
  const __bf16* vbase = Vt + (size_t)((b * HH + h) * DHD) * SS + half * 16;

  v8f o0 = {}, o1 = {}, o2 = {}, o3 = {};
  float mrun[8], lrun[8];
#pragma unroll
  for (int r = 0; r < 8; ++r) { mrun[r] = -1e30f; lrun[r] = 0.f; }
  __bf16* myp = pl + wave * (16 * 32);

  for (int kb2 = 0; kb2 < SS; kb2 += 32) {
    const __bf16* k0 = kbase + (size_t)(kb2 + ln) * DD;        // keys tile 0
    const __bf16* k1 = kbase + (size_t)(kb2 + 16 + ln) * DD;   // keys tile 1
    v8f s0 = {}, s1 = {};
    s0 = wmma_bf16(a0, ld_b_frag(k0), s0);
    s0 = wmma_bf16(a1, ld_b_frag(k0 + 32), s0);
    s1 = wmma_bf16(a0, ld_b_frag(k1), s1);
    s1 = wmma_bf16(a1, ld_b_frag(k1 + 32), s1);

    float p0[8], p1[8];
#pragma unroll
    for (int r = 0; r < 8; ++r) {
      float mx = fmaxf(s0[r], s1[r]);
#pragma unroll
      for (int m = 8; m >= 1; m >>= 1) mx = fmaxf(mx, __shfl_xor(mx, m, 32));
      const float mnew  = fmaxf(mrun[r], mx);
      const float alpha = __expf(mrun[r] - mnew);
      mrun[r] = mnew;
      p0[r] = __expf(s0[r] - mnew);
      p1[r] = __expf(s1[r] - mnew);
      float ls = p0[r] + p1[r];
#pragma unroll
      for (int m = 8; m >= 1; m >>= 1) ls += __shfl_xor(ls, m, 32);
      lrun[r] = lrun[r] * alpha + ls;
      o0[r] *= alpha; o1[r] *= alpha; o2[r] *= alpha; o3[r] *= alpha;
    }
    // C-layout -> A-layout via per-wave LDS patch
#pragma unroll
    for (int r = 0; r < 8; ++r) {
      const int rr = r + 8 * half;
      myp[rr * 32 + ln]      = (__bf16)p0[r];
      myp[rr * 32 + 16 + ln] = (__bf16)p1[r];
    }
    __syncthreads();
    const v16bf pf = ld_a_frag(myp + ln * 32, half);
    __syncthreads();

    const __bf16* v0 = vbase + (size_t)ln * SS + kb2;
    o0 = wmma_bf16(pf, ld_b_frag(v0), o0);
    o1 = wmma_bf16(pf, ld_b_frag(v0 + (size_t)16 * SS), o1);
    o2 = wmma_bf16(pf, ld_b_frag(v0 + (size_t)32 * SS), o2);
    o3 = wmma_bf16(pf, ld_b_frag(v0 + (size_t)48 * SS), o3);
  }

#pragma unroll
  for (int r = 0; r < 8; ++r) {
    const int row = qt * 16 + r + 8 * half;
    const float invl = 1.0f / lrun[r];
    __bf16* dst = ctx + (size_t)(b * SS + row) * DD + h * DHD + ln;
    dst[0]  = (__bf16)(o0[r] * invl);
    dst[16] = (__bf16)(o1[r] * invl);
    dst[32] = (__bf16)(o2[r] * invl);
    dst[48] = (__bf16)(o3[r] * invl);
  }
}

// ---------------------------------------------------------------------------
// Relation net: one WG per (b, chunk, i).  g = gelu(A_i + B_j) in LDS bf16,
// y = g @ W2 + b2 in LDS f32 via WMMA, per-row LN, mean over j, residual.
// ---------------------------------------------------------------------------

__global__ __launch_bounds__(256) void relnet_kernel(
    const float* __restrict__ Abuf, const float* __restrict__ Bbuf,
    const __bf16* __restrict__ W2t, const float* __restrict__ b2,
    const float* __restrict__ lng, const float* __restrict__ lnb,
    const float* __restrict__ x1, float* __restrict__ x2) {
  extern __shared__ char smem[];
  constexpr int GS = 776;  // bf16 row stride (1552 B, 16B aligned)
  constexpr int YS = 772;  // f32 row stride (3088 B, 16B aligned)
  __bf16* g = (__bf16*)smem;                       // 64*GS bf16 = 99328 B
  float*  y = (float*)(smem + 64 * GS * 2);        // 64*YS f32  = 197632 B
  float*  Ai = (float*)(smem + 64 * GS * 2 + 64 * YS * 4);  // 3072 B

  const int tid = threadIdx.x, lane = tid & 31, wave = tid >> 5;
  const int half = lane >> 4, ln = lane & 15;
  const int blk = blockIdx.x;                      // b*1024 + chunk*64 + i
  const int b = blk / (16 * CHC);
  const int rm = blk % (16 * CHC);
  const int chunk = rm / CHC;
  const int i = rm % CHC;
  const int rowbase = b * SS + chunk * CHC;

  for (int c = tid; c < DD; c += 256)
    Ai[c] = Abuf[(size_t)(rowbase + i) * DD + c];
  __syncthreads();

  for (int j = 0; j < CHC; ++j) {
    const float* Bj = Bbuf + (size_t)(rowbase + j) * DD;
    for (int c = tid; c < DD; c += 256)
      g[j * GS + c] = (__bf16)gelu_f(Ai[c] + Bj[c]);
  }
  __syncthreads();

  // GEMM: 192 output tiles (4 x 48), 24 per wave
  for (int tile = wave; tile < 192; tile += 8) {
    const int mt = (tile & 3) * 16, nt = (tile >> 2) * 16;
    const __bf16* arow = g + (mt + ln) * GS;
    const __bf16* brow = W2t + (size_t)(nt + ln) * DD + half * 16;
    v8f acc = {};
    for (int kb = 0; kb < DD; kb += 32)
      acc = wmma_bf16(ld_a_frag(arow + kb, half), ld_b_frag(brow + kb), acc);
    const float bb = b2[nt + ln];
#pragma unroll
    for (int r = 0; r < 8; ++r)
      y[(mt + r + 8 * half) * YS + nt + ln] = acc[r] + bb;
  }
  __syncthreads();

  // LayerNorm each of the 64 rows (8 rows per wave)
  for (int jj = 0; jj < 8; ++jj) {
    const int j = wave * 8 + jj;
    float* yr = y + j * YS;
    float s = 0.f, q = 0.f;
    for (int c = lane; c < DD; c += 32) { float v = yr[c]; s += v; q += v * v; }
#pragma unroll
    for (int m = 16; m >= 1; m >>= 1) {
      s += __shfl_xor(s, m, 32);
      q += __shfl_xor(q, m, 32);
    }
    const float mean = s * (1.0f / DD);
    const float var  = q * (1.0f / DD) - mean * mean;
    const float inv  = rsqrtf(var + 1e-5f);
    for (int c = lane; c < DD; c += 32)
      yr[c] = (yr[c] - mean) * inv * lng[c] + lnb[c];
  }
  __syncthreads();

  // mean over j + residual into x2[rowbase + i]
  for (int c = tid; c < DD; c += 256) {
    float s = 0.f;
    for (int j = 0; j < CHC; ++j) s += y[j * YS + c];
    const size_t o = (size_t)(rowbase + i) * DD + c;
    x2[o] = x1[o] + s * (1.0f / CHC);
  }
}

// ---------------------------------------------------------------------------
// Host driver
// ---------------------------------------------------------------------------

extern "C" void kernel_launch(void* const* d_in, const int* in_sizes, int n_in,
                              void* d_out, int out_size, void* d_ws, size_t ws_size,
                              hipStream_t stream) {
  (void)in_sizes; (void)n_in; (void)out_size; (void)ws_size;
  const float* x      = (const float*)d_in[0];
  const float* Wq     = (const float*)d_in[1];
  const float* bq     = (const float*)d_in[2];
  const float* Wk     = (const float*)d_in[3];
  const float* bk     = (const float*)d_in[4];
  const float* Wv     = (const float*)d_in[5];
  const float* bv     = (const float*)d_in[6];
  const float* Wo     = (const float*)d_in[7];
  const float* bo     = (const float*)d_in[8];
  const float* ln1_g  = (const float*)d_in[9];
  const float* ln1_b  = (const float*)d_in[10];
  const float* rel_W1 = (const float*)d_in[11];
  const float* rel_b1 = (const float*)d_in[12];
  const float* rel_W2 = (const float*)d_in[13];
  const float* rel_b2 = (const float*)d_in[14];
  const float* rln_g  = (const float*)d_in[15];
  const float* rln_b  = (const float*)d_in[16];
  const float* ln2_g  = (const float*)d_in[17];
  const float* ln2_b  = (const float*)d_in[18];
  const float* enc_W1 = (const float*)d_in[19];
  const float* enc_b1 = (const float*)d_in[20];
  const float* enc_W2 = (const float*)d_in[21];
  const float* enc_b2 = (const float*)d_in[22];

  char* ws = (char*)d_ws;
  size_t off = 0;
  auto alloc = [&](size_t bytes) -> void* {
    void* p = ws + off;
    off = (off + bytes + 255) & ~(size_t)255;
    return p;
  };

  __bf16* Wqt    = (__bf16*)alloc((size_t)DD * DD * 2);
  __bf16* Wkt    = (__bf16*)alloc((size_t)DD * DD * 2);
  __bf16* Wvt    = (__bf16*)alloc((size_t)DD * DD * 2);
  __bf16* Wot    = (__bf16*)alloc((size_t)DD * DD * 2);
  __bf16* W1at   = (__bf16*)alloc((size_t)DD * DD * 2);
  __bf16* W1bt   = (__bf16*)alloc((size_t)DD * DD * 2);
  __bf16* W2t    = (__bf16*)alloc((size_t)DD * DD * 2);
  __bf16* encW1t = (__bf16*)alloc((size_t)DD * DFFC * 2);
  __bf16* encW2t = (__bf16*)alloc((size_t)DFFC * DD * 2);
  __bf16* hbf    = (__bf16*)alloc((size_t)NTOK * DD * 2);
  __bf16* qb     = (__bf16*)alloc((size_t)NTOK * DD * 2);
  __bf16* kbuf   = (__bf16*)alloc((size_t)NTOK * DD * 2);
  __bf16* vtb    = (__bf16*)alloc((size_t)NTOK * DD * 2);
  __bf16* ctx    = (__bf16*)alloc((size_t)NTOK * DD * 2);
  __bf16* x1bf   = (__bf16*)alloc((size_t)NTOK * DD * 2);
  __bf16* h2bf   = (__bf16*)alloc((size_t)NTOK * DD * 2);
  __bf16* tbf    = (__bf16*)alloc((size_t)NTOK * DFFC * 2);
  float*  x1     = (float*)alloc((size_t)NTOK * DD * 4);
  float*  Abuf   = (float*)alloc((size_t)NTOK * DD * 4);
  float*  Bbuf   = (float*)alloc((size_t)NTOK * DD * 4);
  float*  x2     = (float*)alloc((size_t)NTOK * DD * 4);

  const dim3 blk(256);
  const int nWW = DD * DD / 256;

  // weights -> transposed bf16
  transpose_cast_kernel<<<nWW, blk, 0, stream>>>(Wq, Wqt, DD, DD);
  transpose_cast_kernel<<<nWW, blk, 0, stream>>>(Wk, Wkt, DD, DD);
  transpose_cast_kernel<<<nWW, blk, 0, stream>>>(Wv, Wvt, DD, DD);
  transpose_cast_kernel<<<nWW, blk, 0, stream>>>(Wo, Wot, DD, DD);
  transpose_cast_kernel<<<nWW, blk, 0, stream>>>(rel_W1, W1at, DD, DD);
  transpose_cast_kernel<<<nWW, blk, 0, stream>>>(rel_W1 + (size_t)DD * DD, W1bt, DD, DD);
  transpose_cast_kernel<<<nWW, blk, 0, stream>>>(rel_W2, W2t, DD, DD);
  transpose_cast_kernel<<<DD * DFFC / 256, blk, 0, stream>>>(enc_W1, encW1t, DD, DFFC);
  transpose_cast_kernel<<<DFFC * DD / 256, blk, 0, stream>>>(enc_W2, encW2t, DFFC, DD);

  // attention block
  ln_rows_bf16<<<NTOK, blk, 0, stream>>>(x, ln1_g, ln1_b, hbf);
  const dim3 g768(NTOK / 64, DD / 64);
  gemm_bf16_kernel<EPI_BF16_QSCALE><<<g768, blk, 0, stream>>>(hbf, Wqt, bq, nullptr, qb,   NTOK, DD, DD);
  gemm_bf16_kernel<EPI_BF16_BIAS  ><<<g768, blk, 0, stream>>>(hbf, Wkt, bk, nullptr, kbuf, NTOK, DD, DD);
  gemm_bf16_kernel<EPI_BF16_VT    ><<<g768, blk, 0, stream>>>(hbf, Wvt, bv, nullptr, vtb,  NTOK, DD, DD);
  attn_kernel<<<BB * HH * (SS / 16) / 8, blk, 0, stream>>>(qb, kbuf, vtb, ctx);
  gemm_bf16_kernel<EPI_F32_BIAS_RES><<<g768, blk, 0, stream>>>(ctx, Wot, bo, x, x1, NTOK, DD, DD);

  // relation net
  cast_bf16_kernel<<<NTOK * DD / 256, blk, 0, stream>>>(x1, x1bf);
  gemm_bf16_kernel<EPI_F32_BIAS><<<g768, blk, 0, stream>>>(x1bf, W1at, rel_b1, nullptr, Abuf, NTOK, DD, DD);
  gemm_bf16_kernel<EPI_F32_NONE><<<g768, blk, 0, stream>>>(x1bf, W1bt, nullptr, nullptr, Bbuf, NTOK, DD, DD);
  const size_t relsmem = (size_t)64 * 776 * 2 + (size_t)64 * 772 * 4 + (size_t)DD * 4; // 300032 B
  relnet_kernel<<<BB * 16 * CHC, blk, relsmem, stream>>>(Abuf, Bbuf, W2t, rel_b2,
                                                         rln_g, rln_b, x1, x2);

  // FFN
  ln_rows_bf16<<<NTOK, blk, 0, stream>>>(x2, ln2_g, ln2_b, h2bf);
  const dim3 gff(NTOK / 64, DFFC / 64);
  gemm_bf16_kernel<EPI_BF16_GELU><<<gff, blk, 0, stream>>>(h2bf, encW1t, enc_b1, nullptr, tbf, NTOK, DFFC, DD);
  gemm_bf16_kernel<EPI_F32_BIAS_RES><<<g768, blk, 0, stream>>>(tbf, encW2t, enc_b2, x2,
                                                               (float*)d_out, NTOK, DD, DFFC);
}
